// ChunkTriangleMultiplicationOutgoing_26637387169906
// MI455X (gfx1250) — compile-verified, matrix operand order checked by reading, not run on hardware
//
#include <hip/hip_runtime.h>

// Triangle multiplication (outgoing), MI455X / gfx1250, wave32 + WMMA f16.
// prep (f32->f16 cast) -> stage1 (LN + gated proj, channel-major f16 out)
// -> tri_mm (128 batched 512^3 GEMMs, 128x128 WG tiles; async global->LDS
// double-buffered pipeline when available, reg-staged fallback otherwise)
// -> stage2 (gate/LN/proj + residual). WMMA operands load as 2x16B per lane.

#define NDIM 512
#define NN   (NDIM * NDIM)
#define DPAIR 128

typedef _Float16 f16;
typedef __attribute__((ext_vector_type(8)))  _Float16 v8h;
typedef __attribute__((ext_vector_type(16))) _Float16 v16h;
typedef __attribute__((ext_vector_type(8)))  float    v8f;
typedef __attribute__((vector_size(16)))     int      v4i_;

#if defined(__has_builtin)
#if __has_builtin(__builtin_amdgcn_global_load_async_to_lds_b128)
#define TRI_HAVE_ASYNC 1
#endif
#endif

#ifdef TRI_HAVE_ASYNC
// 16B async copy global->LDS (GLOBAL_LOAD_ASYNC_TO_LDS_B128, ASYNCcnt).
__device__ __forceinline__ void async_cp16(const f16* g, f16* l) {
  __builtin_amdgcn_global_load_async_to_lds_b128(
      (v4i_*)(const_cast<f16*>(g)), (v4i_*)(l), 0, 0);
}
__device__ __forceinline__ void wait_async0() {
#if defined(__has_builtin) && __has_builtin(__builtin_amdgcn_s_wait_asynccnt)
  __builtin_amdgcn_s_wait_asynccnt(0);
#else
  asm volatile("s_wait_asynccnt 0" ::: "memory");
#endif
}
#endif

// ---- WMMA fragment loader (CDNA5 16-bit A/B striping, ISA 7.12.2) ----
// Row-major [rows][ldk], fragment = 16 rows x 32 k. Lane group g=lane>>4:
// elements 0..7 = k 8g..8g+7 (contiguous), 8..15 = k 16+8g..23+8g (contiguous)
// -> two 16B vector loads per lane. Requires ldk % 8 == 0, 16B-aligned base.
__device__ __forceinline__ v16h frag_rowmajor(const f16* __restrict__ base, int ldk) {
  const int lane = threadIdx.x & 31;
  const int g = lane >> 4, m = lane & 15;
  const f16* p = base + m * ldk + g * 8;
  const v8h lo = *(const v8h*)(p);
  const v8h hi = *(const v8h*)(p + 16);
  return __builtin_shufflevector(lo, hi, 0, 1, 2, 3, 4, 5, 6, 7,
                                 8, 9, 10, 11, 12, 13, 14, 15);
}

__device__ __forceinline__ float sigmoidf_(float x) { return 1.0f / (1.0f + __expf(-x)); }

// ---- Kernel 0: weights -> f16 (native [out][in] row-major layout) ----
__global__ __launch_bounds__(256) void tri_prep_weights(
    const float* __restrict__ lrp_w, const float* __restrict__ gate_w,
    const float* __restrict__ og_w, const float* __restrict__ op_w,
    f16* __restrict__ wcat, f16* __restrict__ ogt, f16* __restrict__ opt) {
  const int idx = blockIdx.x * blockDim.x + threadIdx.x;
  if (idx < 256 * DPAIR) {
    wcat[idx] = (f16)lrp_w[idx];
  } else if (idx < 512 * DPAIR) {
    wcat[idx] = (f16)gate_w[idx - 256 * DPAIR];
  } else if (idx < 512 * DPAIR + DPAIR * DPAIR) {
    const int t = idx - 512 * DPAIR;
    ogt[t] = (f16)og_w[t];
  } else if (idx < 512 * DPAIR + 2 * DPAIR * DPAIR) {
    const int t = idx - 512 * DPAIR - DPAIR * DPAIR;
    opt[t] = (f16)op_w[t];
  }
}

// ---- Kernel 1: LayerNorm + fused gated projections ----
// 64 positions/WG; GEMM [64x128] x [128x512]^T; writes left/right [c][p][q] f16.
__global__ __launch_bounds__(256) void tri_stage1(
    const float* __restrict__ Z, const float* __restrict__ maskr,
    const float* __restrict__ ln1w, const float* __restrict__ ln1b,
    const f16* __restrict__ wcat,
    const float* __restrict__ lrp_b, const float* __restrict__ gate_b,
    f16* __restrict__ leftT, f16* __restrict__ rightT) {
  __shared__ __align__(16) f16 zsh[64 * 136];
  __shared__ __align__(16) f16 xp[8 * 256];   // per-wave 16x16 transpose tile

  const int pos0 = blockIdx.x * 64;
  const int p = pos0 >> 9, q0 = pos0 & 511;
  const int w = threadIdx.x >> 5, lane = threadIdx.x & 31;

  // LayerNorm: wave w handles 8 rows, lane covers 4 features each.
#pragma unroll
  for (int j = 0; j < 8; ++j) {
    const int row = w * 8 + j;
    const int pos = pos0 + row;
    const float4 x = *(const float4*)(Z + (size_t)pos * DPAIR + lane * 4);
    float s1 = x.x + x.y + x.z + x.w;
    float s2 = x.x * x.x + x.y * x.y + x.z * x.z + x.w * x.w;
#pragma unroll
    for (int off = 16; off > 0; off >>= 1) {
      s1 += __shfl_xor(s1, off, 32);
      s2 += __shfl_xor(s2, off, 32);
    }
    const float mean = s1 * (1.0f / 128.0f);
    const float var = s2 * (1.0f / 128.0f) - mean * mean;
    const float rs = rsqrtf(var + 1e-5f);
    const float4 gw = *(const float4*)(ln1w + lane * 4);
    const float4 gb = *(const float4*)(ln1b + lane * 4);
    f16* zr = zsh + row * 136 + lane * 4;
    zr[0] = (f16)((x.x - mean) * rs * gw.x + gb.x);
    zr[1] = (f16)((x.y - mean) * rs * gw.y + gb.y);
    zr[2] = (f16)((x.z - mean) * rs * gw.z + gb.z);
    zr[3] = (f16)((x.w - mean) * rs * gw.w + gb.w);
  }
  __syncthreads();

  // 64 paired (proj,gate) 16x16 output blocks; 8 per wave.
#pragma unroll 1
  for (int it = 0; it < 8; ++it) {
    const int b = it * 8 + w;
    const int tm = b >> 4, tc = b & 15;
    v8f accp = {}, accg = {};
#pragma unroll
    for (int ks = 0; ks < 4; ++ks) {
      const v16h a = frag_rowmajor(zsh + tm * 16 * 136 + ks * 32, 136);
      const v16h bp = frag_rowmajor(wcat + (size_t)(tc * 16) * DPAIR + ks * 32, DPAIR);
      const v16h bg = frag_rowmajor(wcat + (size_t)(256 + tc * 16) * DPAIR + ks * 32, DPAIR);
      accp = __builtin_amdgcn_wmma_f32_16x16x32_f16(false, a, false, bp, (short)0, accp, false, false);
      accg = __builtin_amdgcn_wmma_f32_16x16x32_f16(false, a, false, bg, (short)0, accg, false, false);
    }
    const int g = lane >> 4, nn = lane & 15;
    const int colp = tc * 16 + nn;          // 0..255
    const float bp_ = lrp_b[colp];
    const float bg_ = gate_b[colp];
    f16* xw = xp + w * 256;                 // wave-local; per-wave DS ops in-order
#pragma unroll
    for (int r = 0; r < 8; ++r) {
      const int rowl = r + 8 * g;           // q-local within 16-row tile
      const int pos = pos0 + tm * 16 + rowl;
      const float mk = maskr[pos];
      const float val = (accp[r] + bp_) * mk * sigmoidf_(accg[r] + bg_);
      xw[nn * 16 + rowl] = (f16)val;        // transpose: [channel][q]
    }
    // Flush: 16 channels x 16 q -> 16B contiguous global stores per lane.
    const int cr = lane >> 1, half = lane & 1;
    const uint4 vv = *(const uint4*)(xw + cr * 16 + half * 8);
    const int ch = (tc & 7) * 16 + cr;
    f16* dstb = (tc < 8) ? leftT : rightT;
    *(uint4*)(dstb + (size_t)ch * NN + (size_t)p * NDIM + q0 + tm * 16 + half * 8) = vv;
  }
}

// ---- Kernel 2: triangle einsum, tri[c] = L_c * R_c^T ----
// Grid (4,4,128): 128x128 output tile/WG; wave: 2 A-frags x 4 B-frags = 8 WMMA
// per k-step. Async double-buffered LDS pipeline when the toolchain exposes
// GLOBAL_LOAD_ASYNC_TO_LDS_B128; otherwise reg-staged single buffer.
__global__ __launch_bounds__(256) void tri_mm(
    const f16* __restrict__ leftT, const f16* __restrict__ rightT,
    float* __restrict__ triW) {
  const int i0 = blockIdx.x * 128, j0 = blockIdx.y * 128, c = blockIdx.z;
  const int w = threadIdx.x >> 5, lane = threadIdx.x & 31;
  const f16* Ag = leftT + (size_t)c * NN + (size_t)i0 * NDIM;
  const f16* Bg = rightT + (size_t)c * NN + (size_t)j0 * NDIM;
  const int row = threadIdx.x >> 2, part = threadIdx.x & 3;

  const int tib = (w >> 1) * 2, tjb = (w & 1) * 4;  // wave: (tib..+1) x (tjb..+3)
  v8f acc[2][4] = {{{}, {}, {}, {}}, {{}, {}, {}, {}}};

#ifdef TRI_HAVE_ASYNC
  __shared__ __align__(16) f16 As[2][128 * 40];
  __shared__ __align__(16) f16 Bs[2][128 * 40];
  // Issue one 32-wide k-slab (4x16B per thread) asynchronously into buffer buf.
  auto issue = [&](int k0, int buf) {
    async_cp16(Ag + (size_t)row * NDIM + k0 + part * 8, &As[buf][row * 40 + part * 8]);
    async_cp16(Ag + (size_t)(row + 64) * NDIM + k0 + part * 8, &As[buf][(row + 64) * 40 + part * 8]);
    async_cp16(Bg + (size_t)row * NDIM + k0 + part * 8, &Bs[buf][row * 40 + part * 8]);
    async_cp16(Bg + (size_t)(row + 64) * NDIM + k0 + part * 8, &Bs[buf][(row + 64) * 40 + part * 8]);
  };
  issue(0, 0);
#pragma unroll 1
  for (int it = 0; it < NDIM / 32; ++it) {
    const int cur = it & 1;
    wait_async0();                 // my async deposits for buf[cur] done
    __syncthreads();               // everyone's done -> buf[cur] readable
    if (it + 1 < NDIM / 32) issue((it + 1) * 32, cur ^ 1);  // stream next slab
#pragma unroll
    for (int u = 0; u < 2; ++u) {
      const v16h a = frag_rowmajor(&As[cur][(tib + u) * 16 * 40], 40);
#pragma unroll
      for (int t = 0; t < 4; ++t) {
        const v16h b = frag_rowmajor(&Bs[cur][(tjb + t) * 16 * 40], 40);
        acc[u][t] = __builtin_amdgcn_wmma_f32_16x16x32_f16(false, a, false, b,
                                                           (short)0, acc[u][t],
                                                           false, false);
      }
    }
  }
#else
  __shared__ __align__(16) f16 As[128 * 40];
  __shared__ __align__(16) f16 Bs[128 * 40];
  uint4 av0 = *(const uint4*)(Ag + (size_t)row * NDIM + part * 8);
  uint4 av1 = *(const uint4*)(Ag + (size_t)(row + 64) * NDIM + part * 8);
  uint4 bv0 = *(const uint4*)(Bg + (size_t)row * NDIM + part * 8);
  uint4 bv1 = *(const uint4*)(Bg + (size_t)(row + 64) * NDIM + part * 8);
#pragma unroll 1
  for (int k0 = 0; k0 < NDIM; k0 += 32) {
    __syncthreads();
    *(uint4*)(As + row * 40 + part * 8) = av0;
    *(uint4*)(As + (row + 64) * 40 + part * 8) = av1;
    *(uint4*)(Bs + row * 40 + part * 8) = bv0;
    *(uint4*)(Bs + (row + 64) * 40 + part * 8) = bv1;
    __syncthreads();
    if (k0 + 32 < NDIM) {
      const int kn = k0 + 32;
      av0 = *(const uint4*)(Ag + (size_t)row * NDIM + kn + part * 8);
      av1 = *(const uint4*)(Ag + (size_t)(row + 64) * NDIM + kn + part * 8);
      bv0 = *(const uint4*)(Bg + (size_t)row * NDIM + kn + part * 8);
      bv1 = *(const uint4*)(Bg + (size_t)(row + 64) * NDIM + kn + part * 8);
    }
#pragma unroll
    for (int u = 0; u < 2; ++u) {
      const v16h a = frag_rowmajor(As + (tib + u) * 16 * 40, 40);
#pragma unroll
      for (int t = 0; t < 4; ++t) {
        const v16h b = frag_rowmajor(Bs + (tjb + t) * 16 * 40, 40);
        acc[u][t] = __builtin_amdgcn_wmma_f32_16x16x32_f16(false, a, false, b,
                                                           (short)0, acc[u][t],
                                                           false, false);
      }
    }
  }
#endif

  const int g = lane >> 4, nn = lane & 15;
  float* outb = triW + (size_t)c * NN;        // channel-major: coalesced stores
#pragma unroll
  for (int u = 0; u < 2; ++u) {
#pragma unroll
    for (int t = 0; t < 4; ++t) {
#pragma unroll
      for (int r = 0; r < 8; ++r) {
        const int i = i0 + (tib + u) * 16 + r + 8 * g;
        outb[(size_t)i * NDIM + j0 + (tjb + t) * 16 + nn] = acc[u][t][r];
      }
    }
  }
}

// ---- Kernel 3: output gate + LN + projection + residual ----
// 32 positions/WG; paired go/z2 GEMMs [32x128]x[128x128]^T; [i][j][c] store.
__global__ __launch_bounds__(256) void tri_stage2(
    const float* __restrict__ triW,
    const float* __restrict__ og_b, const float* __restrict__ ln2w,
    const float* __restrict__ ln2b, const float* __restrict__ out_bias,
    const f16* __restrict__ ogt, const f16* __restrict__ opt,
    float* __restrict__ out) {
  __shared__ __align__(16) float ts[32 * 132];
  __shared__ __align__(16) f16 araw[32 * 136];
  __shared__ __align__(16) f16 aln[32 * 136];
  const int pos0 = blockIdx.x * 32;
  const int p = pos0 >> 9, q0 = pos0 & 511;
  const int w = threadIdx.x >> 5, lane = threadIdx.x & 31;

  // Gather 32 positions x 128 channels from channel-major tri (64B runs).
  {
    const int c = threadIdx.x >> 1, half = threadIdx.x & 1;
    const float* src = triW + (size_t)c * NN + (size_t)p * NDIM + q0 + half * 16;
#pragma unroll
    for (int u = 0; u < 4; ++u) {
      const float4 v = *(const float4*)(src + u * 4);
      const int qq = half * 16 + u * 4;
      ts[(qq + 0) * 132 + c] = v.x;
      ts[(qq + 1) * 132 + c] = v.y;
      ts[(qq + 2) * 132 + c] = v.z;
      ts[(qq + 3) * 132 + c] = v.w;
    }
  }
  __syncthreads();

  // LayerNorm over channels; emit raw-f16 (gate GEMM) and LN-f16 (proj GEMM).
#pragma unroll
  for (int j = 0; j < 4; ++j) {
    const int q = w * 4 + j;
    const float4 x = *(const float4*)(ts + q * 132 + lane * 4);
    float s1 = x.x + x.y + x.z + x.w;
    float s2 = x.x * x.x + x.y * x.y + x.z * x.z + x.w * x.w;
#pragma unroll
    for (int off = 16; off > 0; off >>= 1) {
      s1 += __shfl_xor(s1, off, 32);
      s2 += __shfl_xor(s2, off, 32);
    }
    const float mean = s1 * (1.0f / 128.0f);
    const float var = s2 * (1.0f / 128.0f) - mean * mean;
    const float rs = rsqrtf(var + 1e-5f);
    const float4 gw = *(const float4*)(ln2w + lane * 4);
    const float4 gb = *(const float4*)(ln2b + lane * 4);
    f16* r0 = araw + q * 136 + lane * 4;
    f16* r1 = aln + q * 136 + lane * 4;
    r0[0] = (f16)x.x; r0[1] = (f16)x.y; r0[2] = (f16)x.z; r0[3] = (f16)x.w;
    r1[0] = (f16)((x.x - mean) * rs * gw.x + gb.x);
    r1[1] = (f16)((x.y - mean) * rs * gw.y + gb.y);
    r1[2] = (f16)((x.z - mean) * rs * gw.z + gb.z);
    r1[3] = (f16)((x.w - mean) * rs * gw.w + gb.w);
  }
  __syncthreads();

  // 16 paired blocks (tm 0..1, td 0..7); 2 per wave.
#pragma unroll 1
  for (int bb = 0; bb < 2; ++bb) {
    const int blk = w * 2 + bb;
    const int tm = blk >> 3, td = blk & 7;
    v8f accg = {}, accz = {};
#pragma unroll
    for (int ks = 0; ks < 4; ++ks) {
      const v16h a1 = frag_rowmajor(araw + tm * 16 * 136 + ks * 32, 136);
      const v16h b1 = frag_rowmajor(ogt + (size_t)(td * 16) * DPAIR + ks * 32, DPAIR);
      accg = __builtin_amdgcn_wmma_f32_16x16x32_f16(false, a1, false, b1, (short)0, accg, false, false);
      const v16h a2 = frag_rowmajor(aln + tm * 16 * 136 + ks * 32, 136);
      const v16h b2 = frag_rowmajor(opt + (size_t)(td * 16) * DPAIR + ks * 32, DPAIR);
      accz = __builtin_amdgcn_wmma_f32_16x16x32_f16(false, a2, false, b2, (short)0, accz, false, false);
    }
    const int g = lane >> 4, nn = lane & 15;
    const int d = td * 16 + nn;
    const float ogb = og_b[d], ob = out_bias[d];
#pragma unroll
    for (int r = 0; r < 8; ++r) {
      const int q = tm * 16 + r + 8 * g;
      const float tv = ts[q * 132 + d];
      const float go = sigmoidf_(accg[r] + ogb);
      out[(size_t)(pos0 + q) * DPAIR + d] = tv + go * (accz[r] + ob);
    }
  }
}

extern "C" void kernel_launch(void* const* d_in, const int* in_sizes, int n_in,
                              void* d_out, int out_size, void* d_ws, size_t ws_size,
                              hipStream_t stream) {
  const float* Z = (const float*)d_in[0];
  const float* maskr = (const float*)d_in[1];
  const float* ln1w = (const float*)d_in[2];
  const float* ln1b = (const float*)d_in[3];
  const float* lrp_w = (const float*)d_in[4];
  const float* lrp_b = (const float*)d_in[5];
  const float* gate_w = (const float*)d_in[6];
  const float* gate_b = (const float*)d_in[7];
  const float* og_w = (const float*)d_in[8];
  const float* og_b = (const float*)d_in[9];
  const float* ln2w = (const float*)d_in[10];
  const float* ln2b = (const float*)d_in[11];
  const float* op_w = (const float*)d_in[12];
  const float* out_bias = (const float*)d_in[13];

  char* ws = (char*)d_ws;
  f16* wcat = (f16*)(ws + 0);                         // 512*128 f16 = 128 KB
  f16* ogt = (f16*)(ws + 131072);                     // 128*128 f16 = 32 KB
  f16* opt = (f16*)(ws + 163840);                     // 128*128 f16 = 32 KB
  f16* leftT = (f16*)(ws + 196608);                   // 128*512*512 f16 = 64 MB
  f16* rightT = (f16*)(ws + 196608 + 67108864);       // 64 MB
  float* triW = (float*)(ws + 196608 + 2 * 67108864); // 128 MB

  tri_prep_weights<<<384, 256, 0, stream>>>(lrp_w, gate_w, og_w, op_w, wcat, ogt, opt);
  tri_stage1<<<NN / 64, 256, 0, stream>>>(Z, maskr, ln1w, ln1b, wcat, lrp_b, gate_b, leftT, rightT);
  dim3 gtri(NDIM / 128, NDIM / 128, DPAIR);
  tri_mm<<<gtri, 256, 0, stream>>>(leftT, rightT, triW);
  tri_stage2<<<NN / 32, 256, 0, stream>>>(triW, og_b, ln2w, ln2b, out_bias, ogt, opt, (float*)d_out);
}
